// MSAOPM_20194936225871
// MI455X (gfx1250) — matrix operand order repaired and checked by reading, not compiled, not run on hardware
//
#include <hip/hip_runtime.h>
#include <stdint.h>

// ---------------------------------------------------------------------------
// Problem constants (from the reference)
// ---------------------------------------------------------------------------
#define NN   512           // N (batch / contraction dim of the big GEMM)
#define LL   384           // L
#define DD   256           // M_DIM
#define CC   32            // C
#define ZZ   128           // Z_DIM
#define RR   (LL * CC)     // 12288 rows/cols of the big GEMM

// Plain (trivially-constructible) vector types.
typedef __attribute__((ext_vector_type(16))) __bf16    v16bf;
typedef __attribute__((ext_vector_type(8)))  float     v8f;
typedef __attribute__((ext_vector_type(4)))  uint32_t  u32x4;
typedef __attribute__((ext_vector_type(8)))  int       i32x8;
typedef __attribute__((ext_vector_type(4)))  int       i32x4;

static __device__ __forceinline__ unsigned short f2bf(float f) {
    union { float f; uint32_t u; } c; c.f = f;
    uint32_t u = c.u;
    uint32_t r = u + 0x7FFFu + ((u >> 16) & 1u);   // round-to-nearest-even
    return (unsigned short)(r >> 16);
}

static __device__ __forceinline__ v8f vzero8() {
    v8f z = {0.f, 0.f, 0.f, 0.f, 0.f, 0.f, 0.f, 0.f};
    return z;
}

static __device__ __forceinline__ float wave_sum(float s) {
    #pragma unroll
    for (int off = 16; off; off >>= 1) s += __shfl_xor(s, off, 32);
    return s;
}

// ---------------------------------------------------------------------------
// Kernel 1: fused LayerNorm + the two (256 -> 32) linears.
//   Abuf[(i*32+a)*512 + n] = bf16( o(n,i,:) . w2[:,a] + b2[a] )   ("a" in ref)
//   Bbuf[(j*32+b)*512 + n] = bf16( o(n,j,:) . w1[:,b] + b1[b] )   ("b" in ref)
// K (=n) contiguous rows: the 16-bit WMMA A-operand friendly layout.
// ---------------------------------------------------------------------------
__global__ __launch_bounds__(256) void k_ln_linear(
    const float* __restrict__ m,  const float* __restrict__ gamma,
    const float* __restrict__ beta,
    const float* __restrict__ w1, const float* __restrict__ b1,
    const float* __restrict__ w2, const float* __restrict__ b2,
    unsigned short* __restrict__ Abuf, unsigned short* __restrict__ Bbuf) {
    __shared__ float so[8][DD];

    const int wave = threadIdx.x >> 5;
    const int lane = threadIdx.x & 31;
    const int row  = blockIdx.x * 8 + wave;   // row = n*LL + i
    const int n = row / LL;
    const int i = row % LL;

    const float* mr = m + (size_t)row * DD;
    float v[8];
    float s = 0.f;
    #pragma unroll
    for (int u = 0; u < 8; ++u) { v[u] = mr[lane + 32 * u]; s += v[u]; }
    const float mu = wave_sum(s) * (1.0f / (float)DD);

    float vs = 0.f;
    #pragma unroll
    for (int u = 0; u < 8; ++u) { float d = v[u] - mu; vs += d * d; }
    const float var = wave_sum(vs) * (1.0f / (float)DD);
    const float rs  = rsqrtf(var + 1e-5f);

    #pragma unroll
    for (int u = 0; u < 8; ++u) {
        const int d = lane + 32 * u;
        so[wave][d] = (v[u] - mu) * rs * gamma[d] + beta[d];
    }
    __syncthreads();

    float acc1 = 0.f, acc2 = 0.f;
    for (int k = 0; k < DD; ++k) {
        const float o = so[wave][k];              // wave-uniform broadcast
        acc1 = fmaf(o, w1[k * CC + lane], acc1);
        acc2 = fmaf(o, w2[k * CC + lane], acc2);
    }
    Bbuf[(size_t)(i * CC + lane) * NN + n] = f2bf(acc1 + b1[lane]);
    Abuf[(size_t)(i * CC + lane) * NN + n] = f2bf(acc2 + b2[lane]);
}

// ---------------------------------------------------------------------------
// Kernel 1b: pre-pack w3 (1024,128) fp32 into bf16 WMMA-B fragments.
// w3p[(((kk*8 + zt)*32 + lane)*8 + v)] holds K = kk*32 + half*16 + 2v (+K+1)
// for column zt*16 + (lane&15).
// ---------------------------------------------------------------------------
__global__ __launch_bounds__(256) void k_pack_w3(
    const float* __restrict__ w3, uint32_t* __restrict__ w3p) {
    const int idx = blockIdx.x * 256 + threadIdx.x;      // 0 .. 65535
    const int v    = idx & 7;
    const int lane = (idx >> 3) & 31;
    const int zt   = (idx >> 8) & 7;
    const int kk   = idx >> 11;
    const int n16  = lane & 15, half = lane >> 4;
    const int K    = kk * 32 + half * 16 + 2 * v;
    const int col  = zt * 16 + n16;
    const unsigned short e0 = f2bf(w3[(size_t)K * ZZ + col]);
    const unsigned short e1 = f2bf(w3[(size_t)(K + 1) * ZZ + col]);
    w3p[idx] = (uint32_t)e0 | ((uint32_t)e1 << 16);
}

// ---------------------------------------------------------------------------
// Kernel 2: op-GEMM (12288 x 12288 x 512, bf16 WMMA, f32 acc) with the w3
// projection fused in the epilogue. Tile staging is done by the Tensor Data
// Mover (tensor_load_to_lds) with hardware LDS row padding, double buffered,
// tracked by TENSORcnt. Inner loop is pure ds_load_b128 + v_wmma.
// ---------------------------------------------------------------------------
#define LDT 40      // padded LDS row stride (bf16 elems): 64B data + 16B pad
#define LDP 1032    // P row stride (bf16 elems): 2064B rows, 16B aligned

union Frag { v16bf v; u32x4 q[2]; };

// Issue one TDM load of a 128x32 bf16 tile (rows strided 512 elems in global)
// into LDS with 80-byte padded rows. D# per CDNA5 ISA §8:
//  group0: count=1 | lds_addr | global_addr(57b) | type=2
//  group1: data_size=2B, pad_enable, pad_interval=16dw, pad_amount=4dw,
//          tensor_dim0=512, tensor_dim1=12288, tile_dim0=32, tile_dim1=128,
//          tensor_dim0_stride=512
static __device__ __forceinline__ void tdm_load_tile(
    const unsigned short* gsrc, uint32_t lds_off) {
    const uint64_t ga = (uint64_t)(uintptr_t)gsrc;
    u32x4 g0;
    g0.x = 1u;                                            // count=1 (user D#)
    g0.y = lds_off;                                       // lds_addr
    g0.z = (uint32_t)ga;                                  // global_addr lo
    g0.w = ((uint32_t)(ga >> 32) & 0x1FFFFFFu) | (2u << 30); // addr hi | type=2
    i32x8 g1;
    g1[0] = (int)((1u << 16)      // data_size = 2 bytes
                | (1u << 20)      // pad_enable
                | (3u << 22)      // pad_interval: 2^(3+1)=16 dwords (64B row)
                | (3u << 25));    // pad_amount: 4 dwords (16B) -> 80B stride
    g1[1] = (int)((uint32_t)NN << 16);        // tensor_dim0 = 512 (lo16)
    g1[2] = (int)((uint32_t)RR << 16);        // dim0 hi=0 | tensor_dim1 lo16
    g1[3] = (int)(32u << 16);                 // dim1 hi=0 | tile_dim0 = 32
    g1[4] = 128;                              // tile_dim1 = 128, tile_dim2 = 0
    g1[5] = NN;                               // tensor_dim0_stride lo32 = 512
    g1[6] = 0;
    g1[7] = 0;
    const i32x4 z4 = {0, 0, 0, 0};            // groups 2/3 unused (2D tensor)
    const i32x8 z8 = {0, 0, 0, 0, 0, 0, 0, 0};
    __builtin_amdgcn_tensor_load_to_lds(g0, g1, z4, z4, z8, 0);
}

__global__ __launch_bounds__(256) void k_gemm_proj(
    const unsigned short* __restrict__ Abuf,
    const unsigned short* __restrict__ Bbuf,
    const uint32_t* __restrict__ w3p,
    const float* __restrict__ b3,
    float* __restrict__ out) {
    __shared__ union {
        unsigned short stage[2][2][128 * LDT];   // [buf][A=0/B=1][row][k]
        unsigned short sP[16 * LDP];             // [pair p][a*32+b]
    } sm;

    const int tid  = threadIdx.x;
    const int lane = tid & 31, wave = tid >> 5;
    const int wm   = wave >> 1, wn = wave & 1;       // 4x2 wave grid
    const int bm   = blockIdx.y, bn = blockIdx.x;
    const int row0 = bm * 128, col0 = bn * 128;
    const int n16  = lane & 15, half = lane >> 4;

    v8f acc[2][4];
    #pragma unroll
    for (int fm = 0; fm < 2; ++fm)
        #pragma unroll
        for (int fn = 0; fn < 4; ++fn) acc[fm][fn] = vzero8();

    // Prologue: DMA the first K-tile pair into buffer 0.
    if (wave == 0) {
        tdm_load_tile(Abuf + (size_t)row0 * NN,
                      (uint32_t)(uintptr_t)(void*)sm.stage[0][0]);
        tdm_load_tile(Bbuf + (size_t)col0 * NN,
                      (uint32_t)(uintptr_t)(void*)sm.stage[0][1]);
        __builtin_amdgcn_s_wait_tensorcnt(0);
    }
    __syncthreads();

    for (int ks = 0; ks < NN / 32; ++ks) {
        const int cur = ks & 1;
        // DMA next K-tile into the other buffer; overlaps with WMMA below.
        if (wave == 0 && ks + 1 < NN / 32) {
            const int kb1 = (ks + 1) * 32;
            tdm_load_tile(Abuf + (size_t)row0 * NN + kb1,
                          (uint32_t)(uintptr_t)(void*)sm.stage[cur ^ 1][0]);
            tdm_load_tile(Bbuf + (size_t)col0 * NN + kb1,
                          (uint32_t)(uintptr_t)(void*)sm.stage[cur ^ 1][1]);
        }

        const unsigned short* tA = sm.stage[cur][0];
        const unsigned short* tB = sm.stage[cur][1];
        Frag aF[2], bF[4];
        #pragma unroll
        for (int fm = 0; fm < 2; ++fm) {         // A 16x32: lane M=n16, K=16h+s
            const unsigned short* ar = tA + (size_t)(wm * 32 + fm * 16 + n16) * LDT;
            aF[fm].q[0] = *(const u32x4*)(ar + 16 * half);       // K = 16h..16h+7
            aF[fm].q[1] = *(const u32x4*)(ar + 16 * half + 8);   // K = 16h+8..
        }
        #pragma unroll
        for (int fn = 0; fn < 4; ++fn) {         // B 32x16: lane N=n16, K=16h+2v
            const unsigned short* br = tB + (size_t)(wn * 64 + fn * 16 + n16) * LDT + half * 16;
            bF[fn].q[0] = *(const u32x4*)(br);
            bF[fn].q[1] = *(const u32x4*)(br + 8);
        }
        #pragma unroll
        for (int fm = 0; fm < 2; ++fm)
            #pragma unroll
            for (int fn = 0; fn < 4; ++fn)
                acc[fm][fn] = __builtin_amdgcn_wmma_f32_16x16x32_bf16(
                    false, aF[fm].v, false, bF[fn].v,
                    (short)0, acc[fm][fn], false, false);

        // Publish the prefetched buffer to all waves.
        if (wave == 0) __builtin_amdgcn_s_wait_tensorcnt(0);
        __syncthreads();
    }

    // ---- Epilogue 1: scatter op/N (bf16) into P: row p=i_loc*4+j_loc, col a*32+b
    const float invN = 1.0f / (float)NN;
    #pragma unroll
    for (int fm = 0; fm < 2; ++fm) {
        #pragma unroll
        for (int fn = 0; fn < 4; ++fn) {
            const int jl = wn * 2 + (fn >> 1);
            const int b0 = (fn & 1) * 16 + n16;
            const int p  = wm * 4 + jl;
            #pragma unroll
            for (int r = 0; r < 8; ++r) {
                const int a = fm * 16 + r + 8 * half;   // C-layout: M = r + 8*half
                sm.sP[p * LDP + a * 32 + b0] = f2bf(acc[fm][fn][r] * invN);
            }
        }
    }
    __syncthreads();

    // ---- Epilogue 2: Z(16x128) = P(16x1024) @ w3 ; wave owns z-cols zt*16..+15
    const int zt = wave;
    v8f zacc = vzero8();
    #pragma unroll 4
    for (int kk = 0; kk < 32; ++kk) {
        Frag aF, bF;
        const unsigned short* pr = sm.sP + (size_t)n16 * LDP + kk * 32;
        aF.q[0] = *(const u32x4*)(pr + 16 * half);
        aF.q[1] = *(const u32x4*)(pr + 16 * half + 8);
        const uint32_t* wb = w3p + (size_t)(((kk * 8 + zt) * 32 + lane) * 8);
        bF.q[0] = *(const u32x4*)(wb);
        bF.q[1] = *(const u32x4*)(wb + 4);
        zacc = __builtin_amdgcn_wmma_f32_16x16x32_bf16(
            false, aF.v, false, bF.v, (short)0, zacc, false, false);
    }

    const int i_base = bm * 4, j_base = bn * 4;
    const int zc = zt * 16 + n16;
    const float bias = b3[zc];
    #pragma unroll
    for (int r = 0; r < 8; ++r) {
        const int p  = r + 8 * half;
        const int il = p >> 2, jl = p & 3;
        out[((size_t)(i_base + il) * LL + (j_base + jl)) * ZZ + zc] = zacc[r] + bias;
    }
}

// ---------------------------------------------------------------------------
// Launch
// ---------------------------------------------------------------------------
extern "C" void kernel_launch(void* const* d_in, const int* in_sizes, int n_in,
                              void* d_out, int out_size, void* d_ws, size_t ws_size,
                              hipStream_t stream) {
    (void)in_sizes; (void)n_in; (void)out_size; (void)ws_size;
    const float* m     = (const float*)d_in[0];
    const float* gamma = (const float*)d_in[1];
    const float* beta  = (const float*)d_in[2];
    const float* w1    = (const float*)d_in[3];
    const float* b1    = (const float*)d_in[4];
    const float* w2    = (const float*)d_in[5];
    const float* b2    = (const float*)d_in[6];
    const float* w3    = (const float*)d_in[7];
    const float* b3    = (const float*)d_in[8];
    float* out = (float*)d_out;

    // Workspace: Abuf/Bbuf (12288 x 512 bf16 each) + packed w3 fragments.
    const size_t abElems = (size_t)RR * NN;                 // 6,291,456
    unsigned short* Abuf = (unsigned short*)d_ws;
    unsigned short* Bbuf = Abuf + abElems;
    uint32_t*       w3p  = (uint32_t*)(Bbuf + abElems);     // 65536 dwords

    k_ln_linear<<<dim3((NN * LL) / 8), dim3(256), 0, stream>>>(
        m, gamma, beta, w1, b1, w2, b2, Abuf, Bbuf);

    k_pack_w3<<<dim3(256), dim3(256), 0, stream>>>(w3, w3p);

    k_gemm_proj<<<dim3(RR / 128, RR / 128), dim3(256), 0, stream>>>(
        Abuf, Bbuf, w3p, b3, out);
}
